// CIF_12481174962509
// MI455X (gfx1250) — matrix operand order repaired and checked by reading, not compiled, not run on hardware
//
#include <hip/hip_runtime.h>

// ---------------- problem constants (match reference) ----------------
#define B_SZ   16
#define S_LEN  4096
#define C_CH   512
#define KTAP   5
#define T_LEN  1024
#define BETA_F 1.0f
#define EPS_F  1e-4f

// ---------------- vector types for WMMA ----------------
typedef __bf16 bf16x8  __attribute__((ext_vector_type(8)));
typedef __bf16 bf16x16 __attribute__((ext_vector_type(16)));
typedef float  f32x8   __attribute__((ext_vector_type(8)));

// ---------------- GEMM tiling ----------------
#define BM 128
#define BN 128
#define BK 32
#define APAD 8                  // LDS row stride 40 bf16 = 80 B (16B aligned)
#define AROWS (BM + KTAP - 1)   // 132 halo rows

// =====================================================================
// K1: conv_w [C_out, C_in, K] f32  ->  Wt[tap][o][i] bf16 (contiguous i)
// =====================================================================
__global__ __launch_bounds__(256) void k_wconv(const float* __restrict__ w,
                                               __bf16* __restrict__ wt) {
  int idx = blockIdx.x * 256 + threadIdx.x;
  const int total = KTAP * C_CH * C_CH;
  if (idx >= total) return;
  int tap = idx / (C_CH * C_CH);
  int rem = idx - tap * C_CH * C_CH;
  int o = rem / C_CH;
  int i = rem - o * C_CH;
  wt[idx] = (__bf16)w[(size_t)o * C_CH * KTAP + (size_t)i * KTAP + tap];
}

// =====================================================================
// K2: zero out[B,T,C]
// =====================================================================
__global__ __launch_bounds__(256) void k_zero(float4* __restrict__ p, int n4) {
  int i = blockIdx.x * 256 + threadIdx.x;
  if (i < n4) p[i] = make_float4(0.f, 0.f, 0.f, 0.f);
}

// =====================================================================
// K3: conv-as-GEMM with WMMA bf16, double-buffered LDS.
//     h[r,o] = sum_tap sum_i x[b, s+tap-2, i] * Wt[tap][o][i] + bias[o]
//     - halo A tile (132 rows) staged once per K-chunk, reused by all taps
//     - B tiles staged by gfx1250 async global->LDS b128 (ASYNCcnt)
//     - ping/pong buffers: async copies for chunk kc+32 overlap the
//       40-WMMA compute phase of chunk kc; one barrier per chunk
// =====================================================================
__global__ __launch_bounds__(256) void k_conv_gemm(const float* __restrict__ x,
                                                   const __bf16* __restrict__ wt,
                                                   const float* __restrict__ conv_b,
                                                   float* __restrict__ h) {
  __shared__ __align__(16) __bf16 As[2][AROWS][BK + APAD];        // 2 x 10.6 KB
  __shared__ __align__(16) __bf16 Bs[2][KTAP][BN][BK + APAD];     // 2 x 51.2 KB

  const int tid  = threadIdx.x;
  const int lane = tid & 31;
  const int wave = tid >> 5;
  const int wm   = wave & 3;   // M sub-block 0..3 (32 rows each)
  const int wn   = wave >> 2;  // N sub-block 0..1 (64 cols each)

  const int mbase = blockIdx.x * BM;
  const int nbase = blockIdx.y * BN;
  const int bb    = mbase >> 12;          // batch (S = 4096; blocks never straddle)
  const int sblk  = mbase & (S_LEN - 1);  // block start within batch

  f32x8 acc[2][4] = {};

  const int half = lane >> 4;   // K-half selector per ISA fragment layout
  const int l16  = lane & 15;

  // ---- loop-invariant loader indexing ----
  // A loader: chunk c covers row c>>1, 16 cols at (c&1)*16; c in {tid, tid+256}
  const int ar0   = tid >> 1;
  const int ac0   = (tid & 1) * 16;
  const int asp0  = sblk + ar0 - 2;
  const bool av0  = (asp0 >= 0) && (asp0 < S_LEN);
  const float* ag0 = x + ((size_t)(bb << 12) + (size_t)(av0 ? asp0 : 0)) * C_CH + ac0;
  const int ar1   = (tid + 256) >> 1;           // only threads with tid<8 use this
  const int ac1   = (tid & 1) * 16;
  const int asp1  = sblk + ar1 - 2;
  const bool av1  = (tid < 8) && (asp1 >= 0) && (asp1 < S_LEN);
  const float* ag1 = x + ((size_t)(bb << 12) + (size_t)(av1 ? asp1 : 0)) * C_CH + ac1;

  auto stageA = [&](int kc, int buf) {
    {
      __bf16* dst = &As[buf][ar0][ac0];
      if (av0) {
        const float* pf = ag0 + kc;
#pragma unroll
        for (int i = 0; i < 16; ++i) dst[i] = (__bf16)pf[i];
      } else {
#pragma unroll
        for (int i = 0; i < 16; ++i) dst[i] = (__bf16)0.f;
      }
    }
    if (tid < 8) {
      __bf16* dst = &As[buf][ar1][ac1];
      if (av1) {
        const float* pf = ag1 + kc;
#pragma unroll
        for (int i = 0; i < 16; ++i) dst[i] = (__bf16)pf[i];
      } else {
#pragma unroll
        for (int i = 0; i < 16; ++i) dst[i] = (__bf16)0.f;
      }
    }
  };

  // B loader: 5*128*4 = 2560 16-byte chunks, 10 per thread, no divergence.
  auto issueB = [&](int kc, int buf) {
#pragma unroll
    for (int it = 0; it < (KTAP * BN * (BK / 8)) / 256; ++it) {
      const int c   = tid + it * 256;
      const int tap = c >> 9;             // 512 chunks per tap (128 rows * 4)
      const int rem = c & 511;
      const int n   = rem >> 2;
      const int col = (rem & 3) * 8;
      const __bf16* gp = wt + (size_t)tap * C_CH * C_CH +
                         (size_t)(nbase + n) * C_CH + kc + col;
      const unsigned ldsoff = (unsigned)(size_t)&Bs[buf][tap][n][col];
      asm volatile("global_load_async_to_lds_b128 %0, %1, off"
                   :: "v"(ldsoff), "v"(gp) : "memory");
    }
  };

  // ---- prologue: fill buffer 0 ----
  issueB(0, 0);
  stageA(0, 0);
  asm volatile("s_wait_asynccnt 0x0" ::: "memory");
  __syncthreads();

  for (int kc = 0; kc < C_CH; kc += BK) {
    const int cur = (kc >> 5) & 1;
    const int nxt = cur ^ 1;

    // prefetch next chunk into the other buffer (overlaps compute below)
    if (kc + BK < C_CH) {
      issueB(kc + BK, nxt);
      stageA(kc + BK, nxt);
    }

    // ---- compute: 5 taps x (2 M-sub x 4 N-sub) = 40 WMMAs per wave ----
#pragma unroll
    for (int tap = 0; tap < KTAP; ++tap) {
      bf16x16 afrag[2];
#pragma unroll
      for (int ms = 0; ms < 2; ++ms) {
        const __bf16* ap = &As[cur][wm * 32 + ms * 16 + l16 + tap][0];
        bf16x8 lo = *(const bf16x8*)(ap + half * 8);
        bf16x8 hi = *(const bf16x8*)(ap + half * 8 + 16);
        afrag[ms] = __builtin_shufflevector(lo, hi, 0, 1, 2, 3, 4, 5, 6, 7,
                                            8, 9, 10, 11, 12, 13, 14, 15);
      }
#pragma unroll
      for (int ns = 0; ns < 4; ++ns) {
        const __bf16* bp = &Bs[cur][tap][wn * 64 + ns * 16 + l16][0];
        bf16x8 blo = *(const bf16x8*)(bp + half * 8);
        bf16x8 bhi = *(const bf16x8*)(bp + half * 8 + 16);
        bf16x16 bfrag = __builtin_shufflevector(blo, bhi, 0, 1, 2, 3, 4, 5, 6, 7,
                                                8, 9, 10, 11, 12, 13, 14, 15);
#pragma unroll
        for (int ms = 0; ms < 2; ++ms)
          acc[ms][ns] = __builtin_amdgcn_wmma_f32_16x16x32_bf16(
              false, afrag[ms], false, bfrag, (short)0, acc[ms][ns], false, false);
      }
    }

    // next-buffer async copies must have landed before anyone reads them
    asm volatile("s_wait_asynccnt 0x0" ::: "memory");
    __syncthreads();
  }

  // ---- epilogue: D layout (VGPR r -> M = r + 8*(lane/16), N = lane%16) ----
  const int mh = half * 8;
#pragma unroll
  for (int ms = 0; ms < 2; ++ms) {
#pragma unroll
    for (int ns = 0; ns < 4; ++ns) {
      const int col  = nbase + wn * 64 + ns * 16 + l16;
      const float bv = conv_b[col];
      const int  m0  = mbase + wm * 32 + ms * 16 + mh;
#pragma unroll
      for (int r = 0; r < 8; ++r)
        h[(size_t)(m0 + r) * C_CH + col] = acc[ms][ns][r] + bv;
    }
  }
}

// =====================================================================
// K4: per-row LayerNorm -> ReLU -> Linear(C,1) -> sigmoid -> alpha
// =====================================================================
__global__ __launch_bounds__(256) void k_ln_lin_sig(const float* __restrict__ h,
                                                    const unsigned char* __restrict__ pad,
                                                    const float* __restrict__ ln_g,
                                                    const float* __restrict__ ln_b,
                                                    const float* __restrict__ lin_w,
                                                    const float* __restrict__ lin_b,
                                                    float* __restrict__ alpha) {
  const int lane = threadIdx.x & 31;
  const int wave = threadIdx.x >> 5;
  const int row  = blockIdx.x * 8 + wave;
  const float* hp = h + (size_t)row * C_CH;

  float v[16];
  float s = 0.f, s2 = 0.f;
#pragma unroll
  for (int i = 0; i < 16; ++i) {
    v[i] = hp[lane + i * 32];
    s += v[i];
    s2 += v[i] * v[i];
  }
#pragma unroll
  for (int off = 16; off > 0; off >>= 1) {
    s  += __shfl_xor(s,  off, 32);
    s2 += __shfl_xor(s2, off, 32);
  }
  const float mu   = s * (1.f / (float)C_CH);
  const float var  = s2 * (1.f / (float)C_CH) - mu * mu;
  const float rinv = rsqrtf(var + 1e-5f);

  float acc = 0.f;
#pragma unroll
  for (int i = 0; i < 16; ++i) {
    const int ch = lane + i * 32;
    float t = (v[i] - mu) * rinv * ln_g[ch] + ln_b[ch];
    t = fmaxf(t, 0.f);
    acc += t * lin_w[ch];
  }
#pragma unroll
  for (int off = 16; off > 0; off >>= 1) acc += __shfl_xor(acc, off, 32);

  if (lane == 0) {
    const float logit = acc + lin_b[0];
    float a = 1.f / (1.f + expf(-logit));
    a *= (pad[row] ? 1.f : 0.f);
    alpha[row] = a;
  }
}

// =====================================================================
// K5: per-batch scan + fire computation.
// =====================================================================
__global__ __launch_bounds__(1024) void k_cif_scan(const float* __restrict__ alpha,
                                                   const int* __restrict__ tlen,
                                                   float* __restrict__ csum,
                                                   int* __restrict__ ridx,
                                                   int* __restrict__ lidx,
                                                   float* __restrict__ rw,
                                                   float* __restrict__ lw,
                                                   int* __restrict__ ext) {
  const int b   = blockIdx.x;
  const int tid = threadIdx.x;
  __shared__ float sb[1024];
  __shared__ float carry;
  __shared__ float scale_sh;
  if (tid == 0) carry = 0.f;
  __syncthreads();

  for (int ch = 0; ch < 4; ++ch) {
    const int s = ch * 1024 + tid;
    sb[tid] = alpha[b * S_LEN + s];
    __syncthreads();
    for (int off = 1; off < 1024; off <<= 1) {
      float t = (tid >= off) ? sb[tid - off] : 0.f;
      __syncthreads();
      sb[tid] += t;
      __syncthreads();
    }
    const float inc = sb[tid] + carry;
    csum[b * S_LEN + s] = inc;
    __syncthreads();
    if (tid == 1023) carry = inc;
    __syncthreads();
  }

  if (tid == 0) {
    const float total   = carry;
    const float desired = BETA_F * (float)tlen[b] + EPS_F;
    scale_sh = desired / total;
  }
  __syncthreads();
  const float scale = scale_sh;

  for (int ch = 0; ch < 4; ++ch) {
    const int s = ch * 1024 + tid;
    const float a  = alpha[b * S_LEN + s] * scale;
    const float cs = csum[b * S_LEN + s] * scale;
    int r = (int)floorf(cs / BETA_F);
    r = min(max(r, 0), T_LEN);
    int l = 0;
    if (s > 0) {
      const float cp = csum[b * S_LEN + s - 1] * scale;
      l = (int)floorf(cp / BETA_F);
      l = min(max(l, 0), T_LEN);
    }
    const int fire = r - l;
    const int e    = max(fire - 1, 0);
    const float rwv = (fire > 0) ? (cs - (float)r * BETA_F) : 0.f;
    const float lwv = a - rwv - (float)e * BETA_F;
    const int g = b * S_LEN + s;
    ridx[g] = r;
    lidx[g] = l;
    rw[g]   = rwv;
    lw[g]   = lwv;
    ext[g]  = e;
  }
}

// =====================================================================
// K6: scatter-add x into out[B,T,C]. One block per (b,s) frame.
// =====================================================================
__global__ __launch_bounds__(256) void k_cif_scatter(const float* __restrict__ x,
                                                     const int* __restrict__ ridx,
                                                     const int* __restrict__ lidx,
                                                     const float* __restrict__ rw,
                                                     const float* __restrict__ lw,
                                                     const int* __restrict__ ext,
                                                     float* __restrict__ out) {
  const int bs = blockIdx.x;
  const int b  = bs >> 12;
  const int r  = ridx[bs];
  const int l  = lidx[bs];
  const int e  = ext[bs];
  const float rwv = rw[bs];
  const float lwv = lw[bs];
  const int fire  = r - l;

  const float* xp = x + (size_t)bs * C_CH;
  float* ob = out + (size_t)b * T_LEN * C_CH;

  for (int c = threadIdx.x; c < C_CH; c += 256) {
    const float v = xp[c];
    if (fire > 0 && r < T_LEN)
      atomicAdd(&ob[(size_t)r * C_CH + c], rwv * v);
    if (l < T_LEN && lwv != 0.f)
      atomicAdd(&ob[(size_t)l * C_CH + c], lwv * v);
    if (e > 0) {
      const int t = l + 1;
      if (t < T_LEN) atomicAdd(&ob[(size_t)t * C_CH + c], BETA_F * v);
    }
  }
}

// =====================================================================
// host launcher
// =====================================================================
extern "C" void kernel_launch(void* const* d_in, const int* in_sizes, int n_in,
                              void* d_out, int out_size, void* d_ws, size_t ws_size,
                              hipStream_t stream) {
  (void)in_sizes; (void)n_in; (void)out_size; (void)ws_size;

  const float*         x      = (const float*)d_in[0];
  const unsigned char* pad    = (const unsigned char*)d_in[1];   // jax bool: 1 byte/elem
  const int*           tlen   = (const int*)d_in[2];
  const float*         conv_w = (const float*)d_in[4];
  const float*         conv_b = (const float*)d_in[5];
  const float*         ln_g   = (const float*)d_in[6];
  const float*         ln_b   = (const float*)d_in[7];
  const float*         lin_w  = (const float*)d_in[8];
  const float*         lin_b  = (const float*)d_in[9];

  float* out_bt = (float*)d_out;                                 // [B,T,C]
  float* alpha  = (float*)d_out + (size_t)B_SZ * T_LEN * C_CH;   // [B,S]

  char* ws = (char*)d_ws;
  __bf16* wt   = (__bf16*)(ws + 0);                  //  2.62 MB : Wt[tap][o][i]
  float*  h    = (float*)(ws + (4u << 20));          // 134.2 MB : h[B*S, C]
  char*   p1   = ws + (4u << 20) + (size_t)B_SZ * S_LEN * C_CH * sizeof(float);
  float*  csum = (float*)(p1);
  int*    ridx = (int*)(p1 + 1 * (size_t)B_SZ * S_LEN * 4);
  int*    lidx = (int*)(p1 + 2 * (size_t)B_SZ * S_LEN * 4);
  float*  rwv  = (float*)(p1 + 3 * (size_t)B_SZ * S_LEN * 4);
  float*  lwv  = (float*)(p1 + 4 * (size_t)B_SZ * S_LEN * 4);
  int*    extv = (int*)(p1 + 5 * (size_t)B_SZ * S_LEN * 4);

  {
    const int total = KTAP * C_CH * C_CH;
    k_wconv<<<(total + 255) / 256, 256, 0, stream>>>(conv_w, wt);
  }
  {
    const int n4 = (B_SZ * T_LEN * C_CH) / 4;
    k_zero<<<(n4 + 255) / 256, 256, 0, stream>>>((float4*)out_bt, n4);
  }
  {
    dim3 grid((B_SZ * S_LEN) / BM, C_CH / BN);   // 512 x 4
    k_conv_gemm<<<grid, 256, 0, stream>>>(x, wt, conv_b, h);
  }
  {
    k_ln_lin_sig<<<(B_SZ * S_LEN) / 8, 256, 0, stream>>>(h, pad, ln_g, ln_b,
                                                         lin_w, lin_b, alpha);
  }
  {
    k_cif_scan<<<B_SZ, 1024, 0, stream>>>(alpha, tlen, csum, ridx, lidx, rwv, lwv, extv);
  }
  {
    k_cif_scatter<<<B_SZ * S_LEN, 256, 0, stream>>>(x, ridx, lidx, rwv, lwv, extv, out_bt);
  }
}
